// TripletMessagePassingNetwork_24644522344687
// MI455X (gfx1250) — matrix operand order repaired and checked by reading, not compiled, not run on hardware
//
#include <hip/hip_runtime.h>
#include <stdint.h>

// ---------------- problem constants (match reference) ----------------
static constexpr int NN = 40000;   // nodes
static constexpr int EE = 160000;  // edges
static constexpr int GG = 512;     // graphs
static constexpr int HD = 256;     // hidden
static constexpr int HEADS = 3;
static constexpr int TH = HEADS * HD;  // 768
static constexpr int IN_F = 64;
static constexpr int EF = 16;
static constexpr int OUTF = 12;
static constexpr int LAYERS = 3;
static constexpr int STEPS = 6;

// ---------------- types for WMMA / TDM ----------------
typedef __bf16 bf16_t;
typedef __attribute__((ext_vector_type(16))) __bf16 v16bf;
typedef __attribute__((ext_vector_type(8)))  __bf16 v8bf;
typedef __attribute__((ext_vector_type(8)))  float  v8f;
typedef __attribute__((ext_vector_type(4))) unsigned int u32x4;
typedef __attribute__((ext_vector_type(8))) int i32x8;
typedef __attribute__((ext_vector_type(4))) int i32x4;

#if __has_builtin(__builtin_amdgcn_tensor_load_to_lds)
#define HAVE_TDM 1
#if __has_include(<hip/amd_detail/amd_gfx1250_TDM.h>)
#define TDM_ARITY6 1   // therock-10.0 headers -> 6-arg builtin
#endif
#endif

// ---------------- helpers ----------------
__device__ inline float celu1(float x) { return x > 0.f ? x : (__expf(x) - 1.f); }
__device__ inline float sigmoidf_(float x) { return 1.f / (1.f + __expf(-x)); }

// order-preserving float<->uint encoding for atomicMax on floats (any sign)
__device__ inline unsigned encodeOrd(float f) {
    unsigned b = __float_as_uint(f);
    return (b & 0x80000000u) ? ~b : (b | 0x80000000u);
}
__device__ inline float decodeOrd(unsigned u) {
    return (u & 0x80000000u) ? __uint_as_float(u ^ 0x80000000u) : __uint_as_float(~u);
}
static constexpr unsigned ENC_NEG_INF = 0x007FFFFFu;  // encodeOrd(-inf)

static inline int cdiv(int a, int b) { return (a + b - 1) / b; }

// ---------------- small utility kernels ----------------
__global__ void __launch_bounds__(256) fill_u32_kernel(unsigned* p, unsigned v, int n) {
    int i = blockIdx.x * blockDim.x + threadIdx.x;
    if (i < n) p[i] = v;
}

__global__ void __launch_bounds__(256) f32_to_bf16_kernel(const float* __restrict__ s,
                                                          bf16_t* __restrict__ d, int n) {
    int i = blockIdx.x * blockDim.x + threadIdx.x;
    if (i < n) d[i] = (bf16_t)s[i];
}

// src: [K,N] row-major f32 ; dst: [N,K] bf16 (transposed, so column n is contiguous in K)
__global__ void __launch_bounds__(256) transpose_to_bf16_kernel(const float* __restrict__ s,
                                                                bf16_t* __restrict__ d,
                                                                int K, int N) {
    int i = blockIdx.x * blockDim.x + threadIdx.x;
    if (i >= K * N) return;
    int k = i / N, n = i - k * N;
    d[(size_t)n * K + k] = (bf16_t)s[i];
}

// ---------------- WMMA bf16 GEMM with TDM-staged A tile ----------------
// C[M,N] = act( A[M,K](bf16 rowmajor) * Bt[N,K](bf16, B transposed) + bias + resid )
// One 16-row A tile per block, staged into LDS once via the Tensor Data Mover;
// 8 waves per block each own one 16x16 C tile along N.
__global__ void __launch_bounds__(256)
gemm_bf16_wmma(const bf16_t* __restrict__ A, const bf16_t* __restrict__ Bt,
               const float* __restrict__ bias, const float* __restrict__ resid,
               float* __restrict__ C, int M, int N, int K, int act) {
    __shared__ bf16_t ldsA[16 * 768];  // max K = 768 -> 24 KB of the WGP's 320 KB LDS
    const int lane = threadIdx.x & 31;
    const int wave = threadIdx.x >> 5;
    const int tn = (blockIdx.x * 8 + wave) * 16;
    const int tm = blockIdx.y * 16;
    const int half = lane >> 4;   // 0 or 1
    const int l15 = lane & 15;

#if defined(HAVE_TDM)
    if (threadIdx.x < 32) {  // wave 0 issues the DMA; TENSORcnt is per-wave
        uint64_t ga = (uint64_t)(uintptr_t)(A + (size_t)tm * K);   // byte addr of tile start
        uint32_t la = (uint32_t)(uintptr_t)&ldsA[0];               // LDS byte offset
        // ---- D# group 0: count=1, lds_addr, global_addr[56:0], type=2 ----
        u32x4 g0;
        g0[0] = 1u;
        g0[1] = la;
        g0[2] = (uint32_t)ga;
        g0[3] = (uint32_t)((ga >> 32) & 0x01FFFFFFu) | (2u << 30);
        // ---- D# group 1: data_size=2B, 2D tile 16 x K, row stride K ----
        uint32_t td0 = (uint32_t)K;
        i32x8 g1;
        g1[0] = (int)(1u << 16);                   // workgroup_mask=0 | data_size=1 (2B)
        g1[1] = (int)((td0 & 0xFFFFu) << 16);      // tensor_dim0[15:0] at bits 63:48
        g1[2] = (int)((td0 >> 16) | (16u << 16));  // tensor_dim0[31:16] | tensor_dim1=16
        g1[3] = (int)(td0 << 16);                  // tensor_dim1 hi=0 | tile_dim0=K
        g1[4] = (int)16u;                          // tile_dim1=16 | tile_dim2=0
        g1[5] = (int)td0;                          // tensor_dim0_stride[31:0]=K
        g1[6] = 0;                                 // stride0 hi | dim1_stride lo
        g1[7] = 0;
        i32x4 z4 = {0, 0, 0, 0};
#if defined(TDM_ARITY6)
        i32x8 z8 = {0, 0, 0, 0, 0, 0, 0, 0};
        __builtin_amdgcn_tensor_load_to_lds(g0, g1, z4, z4, z8, 0);
#else
        __builtin_amdgcn_tensor_load_to_lds(g0, g1, z4, z4, 0);
#endif
        __builtin_amdgcn_s_wait_tensorcnt(0);
    }
    __syncthreads();
#else
    // fallback: cooperative copy of the A tile into LDS
    for (int i = threadIdx.x; i < 16 * (K >> 3); i += 256) {
        int r = i / (K >> 3), c = i - r * (K >> 3);
        *(v8bf*)(ldsA + (size_t)r * K + c * 8) =
            *(const v8bf*)(A + (size_t)(tm + r) * K + c * 8);
    }
    __syncthreads();
#endif

    const bf16_t* aRow = ldsA + (size_t)l15 * K;       // A row for M = lane%16 (from LDS)
    const bf16_t* bRow = Bt + (size_t)(tn + l15) * K;  // B column for N = lane%16

    v8f acc = {};
    for (int k0 = 0; k0 < K; k0 += 32) {
        // A 16x32 bf16 layout: lanes 0-15 elems[0..7]=K 0..7, [8..15]=K 16..23;
        //                      lanes16-31 elems[0..7]=K 8..15, [8..15]=K 24..31
        v8bf a0 = *(const v8bf*)(aRow + k0 + half * 8);
        v8bf a1 = *(const v8bf*)(aRow + k0 + 16 + half * 8);
        v16bf a;
#pragma unroll
        for (int i = 0; i < 8; ++i) { a[i] = a0[i]; a[i + 8] = a1[i]; }
        // B 32x16 bf16: same K layout as A (lanes 0-15 K 0..7,16..23; lanes 16-31 K 8..15,24..31)
        v8bf b0 = *(const v8bf*)(bRow + k0 + half * 8);
        v8bf b1 = *(const v8bf*)(bRow + k0 + 16 + half * 8);
        v16bf b;
#pragma unroll
        for (int i = 0; i < 8; ++i) { b[i] = b0[i]; b[i + 8] = b1[i]; }
        acc = __builtin_amdgcn_wmma_f32_16x16x32_bf16(false, a, false, b,
                                                      (short)0, acc, false, false);
    }
    // C/D layout: VGPR v -> M = v + half*8 ; N = lane%16
#pragma unroll
    for (int v = 0; v < 8; ++v) {
        int row = tm + half * 8 + v;
        int col = tn + l15;
        float val = acc[v];
        if (bias)  val += bias[col];
        if (resid) val += resid[(size_t)row * N + col];
        if (act)   val = celu1(val);
        C[(size_t)row * N + col] = val;
    }
}

// ---------------- edge kernels (one wave of 32 lanes per edge) ----------------
// Pass 1: attention logits (leaky-relu) + per-(dst,head) running max
__global__ void __launch_bounds__(256)
edge_alpha_kernel(const float* __restrict__ xn, const float* __restrict__ eattr,
                  const float* __restrict__ We_l, const float* __restrict__ Watt_l,
                  const int* __restrict__ src, const int* __restrict__ dst,
                  float* __restrict__ alphaB, unsigned* __restrict__ mEnc, int nE) {
    int wid = (blockIdx.x * blockDim.x + threadIdx.x) >> 5;
    int lane = threadIdx.x & 31;
    if (wid >= nE) return;
    int s = src[wid], d = dst[wid];
    float ea16[EF];
#pragma unroll
    for (int k = 0; k < EF; ++k) ea16[k] = eattr[(size_t)wid * EF + k];
    const float* xi = xn + (size_t)d * TH;
    const float* xj = xn + (size_t)s * TH;
    float acc[HEADS];
#pragma unroll
    for (int h = 0; h < HEADS; ++h) {
        float a = 0.f;
        const float* wa = Watt_l + h * TH;  // [0:256)=x_i, [256:512)=ea, [512:768)=x_j
#pragma unroll
        for (int j = 0; j < 8; ++j) {
            int dd = j * 32 + lane;        // 0..255 within head
            int col = h * HD + dd;         // 0..767
            float eav = 0.f;
#pragma unroll
            for (int k = 0; k < EF; ++k) eav += ea16[k] * We_l[k * TH + col];
            a += xi[col] * wa[dd] + eav * wa[HD + dd] + xj[col] * wa[2 * HD + dd];
        }
        acc[h] = a;
    }
#pragma unroll
    for (int h = 0; h < HEADS; ++h)
#pragma unroll
        for (int off = 16; off; off >>= 1) acc[h] += __shfl_xor(acc[h], off, 32);
    if (lane == 0) {
#pragma unroll
        for (int h = 0; h < HEADS; ++h) {
            float a = acc[h];
            a = a > 0.f ? a : 0.2f * a;  // leaky relu
            alphaB[(size_t)wid * HEADS + h] = a;
            atomicMax(&mEnc[(size_t)d * HEADS + h], encodeOrd(a));
        }
    }
}

// Pass 2: exp(alpha - max) and per-(dst,head) sum
__global__ void __launch_bounds__(256)
edge_exp_kernel(float* __restrict__ alphaB, const int* __restrict__ dst,
                const unsigned* __restrict__ mEnc, float* __restrict__ sSum, int nE) {
    int i = blockIdx.x * blockDim.x + threadIdx.x;
    if (i >= nE * HEADS) return;
    int e = i / HEADS, h = i - e * HEADS;
    int d = dst[e];
    float m = decodeOrd(mEnc[(size_t)d * HEADS + h]);
    float ex = __expf(alphaB[i] - m);
    alphaB[i] = ex;
    atomicAdd(&sSum[(size_t)d * HEADS + h], ex);
}

// Pass 3: msg = softmax * ea * x_j, scatter-add into aggr[dst]
__global__ void __launch_bounds__(256)
edge_msg_kernel(const float* __restrict__ xn, const float* __restrict__ eattr,
                const float* __restrict__ We_l,
                const int* __restrict__ src, const int* __restrict__ dst,
                const float* __restrict__ alphaB, const float* __restrict__ sSum,
                float* __restrict__ aggr, int nE) {
    int wid = (blockIdx.x * blockDim.x + threadIdx.x) >> 5;
    int lane = threadIdx.x & 31;
    if (wid >= nE) return;
    int s = src[wid], d = dst[wid];
    float ea16[EF];
#pragma unroll
    for (int k = 0; k < EF; ++k) ea16[k] = eattr[(size_t)wid * EF + k];
    const float* xj = xn + (size_t)s * TH;
    float* ag = aggr + (size_t)d * TH;
    float a[HEADS];
#pragma unroll
    for (int h = 0; h < HEADS; ++h)
        a[h] = alphaB[(size_t)wid * HEADS + h] / (sSum[(size_t)d * HEADS + h] + 1e-16f);
#pragma unroll
    for (int h = 0; h < HEADS; ++h) {
#pragma unroll
        for (int j = 0; j < 8; ++j) {
            int col = h * HD + j * 32 + lane;
            float eav = 0.f;
#pragma unroll
            for (int k = 0; k < EF; ++k) eav += ea16[k] * We_l[k * TH + col];
            atomicAdd(&ag[col], a[h] * eav * xj[col]);
        }
    }
}

// ---------------- Set2Set kernels ----------------
__global__ void __launch_bounds__(256)
s2s_gates_kernel(const float* __restrict__ q, const float* __restrict__ h,
                 const float* __restrict__ Wih, const float* __restrict__ Whh,
                 const float* __restrict__ bih, const float* __restrict__ bhh,
                 float* __restrict__ gates) {
    int i = blockIdx.x * blockDim.x + threadIdx.x;
    if (i >= GG * 4 * HD) return;
    int g = i >> 10, r = i & 1023;
    float v = bih[r] + bhh[r];
    const float* qr = q + (size_t)g * 2 * HD;
    const float* wr = Wih + (size_t)r * 2 * HD;
    for (int k = 0; k < 2 * HD; ++k) v += qr[k] * wr[k];
    const float* hr = h + (size_t)g * HD;
    const float* w2 = Whh + (size_t)r * HD;
    for (int k = 0; k < HD; ++k) v += hr[k] * w2[k];
    gates[i] = v;
}

__global__ void __launch_bounds__(256)
s2s_cell_kernel(const float* __restrict__ gates, float* __restrict__ cG,
                float* __restrict__ hG) {
    int i = blockIdx.x * blockDim.x + threadIdx.x;
    if (i >= GG * HD) return;
    int g = i >> 8, d = i & 255;
    const float* gt = gates + (size_t)g * 4 * HD;
    float ig = sigmoidf_(gt[d]);
    float fg = sigmoidf_(gt[HD + d]);
    float gg = tanhf(gt[2 * HD + d]);
    float og = sigmoidf_(gt[3 * HD + d]);
    float c = fg * cG[i] + ig * gg;
    cG[i] = c;
    hG[i] = og * tanhf(c);
}

// e[n] = dot(out[n], h[batch[n]]); atomic-max per graph
__global__ void __launch_bounds__(256)
s2s_e_kernel(const float* __restrict__ out, const float* __restrict__ hG,
             const int* __restrict__ batch, float* __restrict__ eN,
             unsigned* __restrict__ mG, int n) {
    int wid = (blockIdx.x * blockDim.x + threadIdx.x) >> 5;
    int lane = threadIdx.x & 31;
    if (wid >= n) return;
    int b = batch[wid];
    const float* o = out + (size_t)wid * HD;
    const float* h = hG + (size_t)b * HD;
    float a = 0.f;
#pragma unroll
    for (int t = 0; t < 8; ++t) { int c = lane + 32 * t; a += o[c] * h[c]; }
#pragma unroll
    for (int off = 16; off; off >>= 1) a += __shfl_xor(a, off, 32);
    if (lane == 0) {
        eN[wid] = a;
        atomicMax(&mG[b], encodeOrd(a));
    }
}

__global__ void __launch_bounds__(256)
s2s_exp_kernel(float* __restrict__ eN, const int* __restrict__ batch,
               const unsigned* __restrict__ mG, float* __restrict__ sG, int n) {
    int i = blockIdx.x * blockDim.x + threadIdx.x;
    if (i >= n) return;
    int b = batch[i];
    float ex = __expf(eN[i] - decodeOrd(mG[b]));
    eN[i] = ex;
    atomicAdd(&sG[b], ex);
}

__global__ void __launch_bounds__(256)
s2s_r_kernel(const float* __restrict__ out, const float* __restrict__ eN,
             const int* __restrict__ batch, const float* __restrict__ sG,
             float* __restrict__ rG, int n) {
    int wid = (blockIdx.x * blockDim.x + threadIdx.x) >> 5;
    int lane = threadIdx.x & 31;
    if (wid >= n) return;
    int b = batch[wid];
    float a = eN[wid] / (sG[b] + 1e-16f);
    const float* o = out + (size_t)wid * HD;
    float* r = rG + (size_t)b * HD;
#pragma unroll
    for (int t = 0; t < 8; ++t) { int c = lane + 32 * t; atomicAdd(&r[c], a * o[c]); }
}

__global__ void __launch_bounds__(256)
s2s_qstar_kernel(const float* __restrict__ hG, const float* __restrict__ rG,
                 float* __restrict__ q) {
    int i = blockIdx.x * blockDim.x + threadIdx.x;
    if (i >= GG * 2 * HD) return;
    int g = i / (2 * HD), k = i - g * 2 * HD;
    q[i] = (k < HD) ? hG[(size_t)g * HD + k] : rG[(size_t)g * HD + (k - HD)];
}

// ---------------- MLP head ----------------
__global__ void __launch_bounds__(256)
mlp1_kernel(const float* __restrict__ q, const float* __restrict__ W1,
            const float* __restrict__ b1, float* __restrict__ y) {
    int i = blockIdx.x * blockDim.x + threadIdx.x;
    if (i >= GG * HD) return;
    int g = i >> 8, d = i & 255;
    float v = b1[d];
    const float* qr = q + (size_t)g * 2 * HD;
    for (int k = 0; k < 2 * HD; ++k) v += qr[k] * W1[(size_t)k * HD + d];
    y[i] = v;
}

__global__ void __launch_bounds__(256)
mlp_ln_relu_kernel(const float* __restrict__ y, const float* __restrict__ g,
                   const float* __restrict__ b, float* __restrict__ yo) {
    __shared__ float s1[HD], s2[HD];
    int gi = blockIdx.x, t = threadIdx.x;
    float v = y[(size_t)gi * HD + t];
    s1[t] = v; s2[t] = v * v;
    __syncthreads();
    for (int st = HD / 2; st; st >>= 1) {
        if (t < st) { s1[t] += s1[t + st]; s2[t] += s2[t + st]; }
        __syncthreads();
    }
    float mu = s1[0] * (1.f / HD);
    float var = s2[0] * (1.f / HD) - mu * mu;
    float yn = (v - mu) * rsqrtf(var + 1e-5f) * g[t] + b[t];
    yo[(size_t)gi * HD + t] = yn > 0.f ? yn : 0.f;
}

__global__ void __launch_bounds__(256)
mlp2_kernel(const float* __restrict__ y, const float* __restrict__ W2,
            const float* __restrict__ b2, float* __restrict__ out) {
    int i = blockIdx.x * blockDim.x + threadIdx.x;
    if (i >= GG * OUTF) return;
    int g = i / OUTF, o = i - g * OUTF;
    float v = b2[o];
    const float* yr = y + (size_t)g * HD;
    for (int k = 0; k < HD; ++k) v += yr[k] * W2[(size_t)k * OUTF + o];
    out[i] = v;
}

// ---------------- host orchestration ----------------
extern "C" void kernel_launch(void* const* d_in, const int* in_sizes, int n_in,
                              void* d_out, int out_size, void* d_ws, size_t ws_size,
                              hipStream_t stream) {
    (void)in_sizes; (void)n_in; (void)out_size; (void)ws_size;
    const float* x         = (const float*)d_in[0];
    const float* edge_attr = (const float*)d_in[1];
    const int*   eidx      = (const int*)d_in[2];
    const int*   batch     = (const int*)d_in[3];
    const float* proj_W    = (const float*)d_in[4];
    const float* proj_b    = (const float*)d_in[5];
    const float* Wn        = (const float*)d_in[6];
    const float* We        = (const float*)d_in[7];
    const float* Watt      = (const float*)d_in[8];
    const float* Ws        = (const float*)d_in[9];
    const float* bl        = (const float*)d_in[10];
    const float* W_ih      = (const float*)d_in[11];
    const float* W_hh      = (const float*)d_in[12];
    const float* b_ih      = (const float*)d_in[13];
    const float* b_hh      = (const float*)d_in[14];
    const float* mlp_W1    = (const float*)d_in[15];
    const float* mlp_b1    = (const float*)d_in[16];
    const float* ln_g      = (const float*)d_in[17];
    const float* ln_b      = (const float*)d_in[18];
    const float* mlp_W2    = (const float*)d_in[19];
    const float* mlp_b2    = (const float*)d_in[20];
    const int* srcI = eidx;        // messages flow src -> dst
    const int* dstI = eidx + EE;
    float* dout = (float*)d_out;

    // workspace layout (256B aligned slices)
    char* ws = (char*)d_ws;
    size_t off = 0;
    auto take = [&](size_t bytes) -> char* {
        off = (off + 255) & ~(size_t)255;
        char* p = ws + off;
        off += bytes;
        return p;
    };
    float*    outF   = (float*)take((size_t)NN * HD * 4);
    float*    xnF    = (float*)take((size_t)NN * TH * 4);
    float*    aggrF  = (float*)take((size_t)NN * TH * 4);
    bf16_t*   Abf    = (bf16_t*)take((size_t)NN * TH * 2);
    bf16_t*   WtB    = (bf16_t*)take((size_t)TH * HD * 2);
    float*    alphaB = (float*)take((size_t)EE * HEADS * 4);
    unsigned* mEnc   = (unsigned*)take((size_t)NN * HEADS * 4);
    float*    sSum   = (float*)take((size_t)NN * HEADS * 4);
    float*    hG     = (float*)take((size_t)GG * HD * 4);
    float*    cG     = (float*)take((size_t)GG * HD * 4);
    float*    qstar  = (float*)take((size_t)GG * 2 * HD * 4);
    float*    rG     = (float*)take((size_t)GG * HD * 4);
    float*    gates  = (float*)take((size_t)GG * 4 * HD * 4);  // also reused as MLP y
    float*    eN     = (float*)take((size_t)NN * 4);
    unsigned* mG     = (unsigned*)take((size_t)GG * 4);
    float*    sG     = (float*)take((size_t)GG * 4);
    float*    yG     = (float*)take((size_t)GG * HD * 4);

    // ---- projection: out = celu(x @ proj_W + proj_b) ----
    f32_to_bf16_kernel<<<cdiv(NN * IN_F, 256), 256, 0, stream>>>(x, Abf, NN * IN_F);
    transpose_to_bf16_kernel<<<cdiv(IN_F * HD, 256), 256, 0, stream>>>(proj_W, WtB, IN_F, HD);
    gemm_bf16_wmma<<<dim3(HD / 128, NN / 16), 256, 0, stream>>>(
        Abf, WtB, proj_b, nullptr, outF, NN, HD, IN_F, /*celu*/1);
    f32_to_bf16_kernel<<<cdiv(NN * HD, 256), 256, 0, stream>>>(outF, Abf, NN * HD);

    // ---- message-passing layers ----
    for (int l = 0; l < LAYERS; ++l) {
        const float* We_l   = We + (size_t)l * EF * TH;
        const float* Watt_l = Watt + (size_t)l * HEADS * TH;

        // xn = out @ Wn[l]   [N,256]x[256,768]
        transpose_to_bf16_kernel<<<cdiv(HD * TH, 256), 256, 0, stream>>>(
            Wn + (size_t)l * HD * TH, WtB, HD, TH);
        gemm_bf16_wmma<<<dim3(TH / 128, NN / 16), 256, 0, stream>>>(
            Abf, WtB, nullptr, nullptr, xnF, NN, TH, HD, 0);

        hipMemsetAsync(aggrF, 0, (size_t)NN * TH * 4, stream);
        hipMemsetAsync(sSum, 0, (size_t)NN * HEADS * 4, stream);
        fill_u32_kernel<<<cdiv(NN * HEADS, 256), 256, 0, stream>>>(mEnc, ENC_NEG_INF, NN * HEADS);

        edge_alpha_kernel<<<cdiv(EE * 32, 256), 256, 0, stream>>>(
            xnF, edge_attr, We_l, Watt_l, srcI, dstI, alphaB, mEnc, EE);
        edge_exp_kernel<<<cdiv(EE * HEADS, 256), 256, 0, stream>>>(
            alphaB, dstI, mEnc, sSum, EE);
        edge_msg_kernel<<<cdiv(EE * 32, 256), 256, 0, stream>>>(
            xnF, edge_attr, We_l, srcI, dstI, alphaB, sSum, aggrF, EE);

        // out = out + aggr @ Ws[l] + bl[l]   [N,768]x[768,256]
        f32_to_bf16_kernel<<<cdiv(NN * TH, 256), 256, 0, stream>>>(aggrF, Abf, NN * TH);
        transpose_to_bf16_kernel<<<cdiv(TH * HD, 256), 256, 0, stream>>>(
            Ws + (size_t)l * TH * HD, WtB, TH, HD);
        gemm_bf16_wmma<<<dim3(HD / 128, NN / 16), 256, 0, stream>>>(
            Abf, WtB, bl + (size_t)l * HD, outF, outF, NN, HD, TH, 0);
        f32_to_bf16_kernel<<<cdiv(NN * HD, 256), 256, 0, stream>>>(outF, Abf, NN * HD);
    }

    // ---- Set2Set pooling ----
    hipMemsetAsync(hG, 0, (size_t)GG * HD * 4, stream);
    hipMemsetAsync(cG, 0, (size_t)GG * HD * 4, stream);
    hipMemsetAsync(qstar, 0, (size_t)GG * 2 * HD * 4, stream);
    for (int s = 0; s < STEPS; ++s) {
        s2s_gates_kernel<<<cdiv(GG * 4 * HD, 256), 256, 0, stream>>>(
            qstar, hG, W_ih, W_hh, b_ih, b_hh, gates);
        s2s_cell_kernel<<<cdiv(GG * HD, 256), 256, 0, stream>>>(gates, cG, hG);
        fill_u32_kernel<<<cdiv(GG, 256), 256, 0, stream>>>(mG, ENC_NEG_INF, GG);
        hipMemsetAsync(sG, 0, (size_t)GG * 4, stream);
        hipMemsetAsync(rG, 0, (size_t)GG * HD * 4, stream);
        s2s_e_kernel<<<cdiv(NN * 32, 256), 256, 0, stream>>>(outF, hG, batch, eN, mG, NN);
        s2s_exp_kernel<<<cdiv(NN, 256), 256, 0, stream>>>(eN, batch, mG, sG, NN);
        s2s_r_kernel<<<cdiv(NN * 32, 256), 256, 0, stream>>>(outF, eN, batch, sG, rG, NN);
        s2s_qstar_kernel<<<cdiv(GG * 2 * HD, 256), 256, 0, stream>>>(hG, rG, qstar);
    }

    // ---- MLP head: Linear -> LayerNorm -> ReLU -> Linear ----
    mlp1_kernel<<<cdiv(GG * HD, 256), 256, 0, stream>>>(qstar, mlp_W1, mlp_b1, gates);
    mlp_ln_relu_kernel<<<GG, HD, 0, stream>>>(gates, ln_g, ln_b, yG);
    mlp2_kernel<<<cdiv(GG * OUTF, 256), 256, 0, stream>>>(yG, mlp_W2, mlp_b2, dout);
}
